// CurveGrouping_8521215115542
// MI455X (gfx1250) — compile-verified, matrix operand order checked by reading, not run on hardware
//
#include <hip/hip_runtime.h>
#include <hip/hip_bf16.h>
#include <math.h>

#define BN 8
#define C 64
#define NPTS 16384
#define KNB 32
#define CN 256
#define CL 32
#define PTOT (BN*NPTS)      /* 131072 */
#define NCURVE (BN*CN)      /* 2048 */
#define EPS_BN 1e-5f

typedef __attribute__((ext_vector_type(2)))  float    v2f;
typedef __attribute__((ext_vector_type(8)))  float    v8f;
typedef __attribute__((ext_vector_type(16))) _Float16 v16h;

#if __has_builtin(__builtin_amdgcn_wmma_f32_16x16x4_f32)
#define HAVE_WMMA_F32X4 1
#elif __has_builtin(__builtin_amdgcn_wmma_f32_16x16x32_f16)
#define HAVE_WMMA_F16 1
#endif

// ---------------- device helpers ----------------

__device__ __forceinline__ float wave_sum32(float v) {
#pragma unroll
  for (int m = 16; m >= 1; m >>= 1) v += __shfl_xor(v, m, 32);
  return v;
}

__device__ __forceinline__ void atomAddF(float* p, float v) {
  __hip_atomic_fetch_add(p, v, __ATOMIC_RELAXED, __HIP_MEMORY_SCOPE_AGENT);
}

__device__ __forceinline__ float ldAgent(float* p) {
  return __hip_atomic_load(p, __ATOMIC_RELAXED, __HIP_MEMORY_SCOPE_AGENT);
}

// Monotonic-counter grid barrier (all blocks must be co-resident).
__device__ __forceinline__ void grid_sync(unsigned int* bar, unsigned int target) {
  __threadfence();
  __syncthreads();
  if (threadIdx.x == 0) {
    __hip_atomic_fetch_add(bar, 1u, __ATOMIC_ACQ_REL, __HIP_MEMORY_SCOPE_AGENT);
    while (__hip_atomic_load(bar, __ATOMIC_ACQUIRE, __HIP_MEMORY_SCOPE_AGENT) < target) {
      __builtin_amdgcn_s_sleep(1);
    }
  }
  __syncthreads();
}

// ---------------- kernels ----------------

__global__ void init_kernel(float* lacc, float* macc, unsigned int* bar) {
  int t = threadIdx.x;
  if (t < CL * 2) lacc[t] = 0.0f;
  if (t < CL * 4) macc[t] = 0.0f;
  if (t == 0) bar[0] = 0u;
}

// x_att[b,n] = sigmoid( sum_c x[b,c,n] * att_w[c] )  via V_WMMA_F32_16X16X4_F32.
// One wave -> 16 points (all in same batch since NPTS%16==0).
__global__ __launch_bounds__(256) void att_kernel(const float* __restrict__ x,
                                                  const float* __restrict__ att_w,
                                                  float* __restrict__ x_att) {
  int wave = (int)((blockIdx.x * blockDim.x + threadIdx.x) >> 5);
  int lane = (int)(threadIdx.x & 31);
  int pbase = wave * 16;
  int b = pbase / NPTS;
  int n0 = pbase % NPTS;
  const float* xb = x + (size_t)b * C * NPTS + n0;
  int row = lane & 15;
  int half = lane >> 4;

#if defined(HAVE_WMMA_F32X4) || defined(HAVE_WMMA_F16)
  v8f acc = {};
#if defined(HAVE_WMMA_F32X4)
  // A: 16x4 f32 tile of x^T (lanes 0-15: K=0,1 ; lanes 16-31: K=2,3)
  // B: 4x16, att_w chunk broadcast across N
#pragma unroll
  for (int kk = 0; kk < 16; ++kk) {
    int c0 = kk * 4 + half * 2;
    v2f A = { xb[(size_t)c0 * NPTS + row], xb[(size_t)(c0 + 1) * NPTS + row] };
    v2f Bm = { att_w[c0], att_w[c0 + 1] };
    acc = __builtin_amdgcn_wmma_f32_16x16x4_f32(false, A, false, Bm,
                                                (short)0, acc, false, false);
  }
#else
  // f16 fallback: two 16x16x32 WMMAs cover the 64-channel K dim.
  union H16 { v16h v; _Float16 e[16]; };
#pragma unroll
  for (int kk = 0; kk < 2; ++kk) {
    int c0 = kk * 32;
    H16 A, Bm;
#pragma unroll
    for (int e = 0; e < 16; ++e) {
      int ka = e + (e >= 8 ? 8 : 0) + half * 8;   // A 16-bit layout
      int kb = e + half * 16;                     // B: contiguous k order, 16*half + e
      A.e[e]  = (_Float16)xb[(size_t)(c0 + ka) * NPTS + row];
      Bm.e[e] = (_Float16)att_w[c0 + kb];
    }
    acc = __builtin_amdgcn_wmma_f32_16x16x32_f16(false, A.v, false, Bm.v,
                                                 (short)0, acc, false, false);
  }
#endif
  // C layout: lane 0 holds rows 0..7 (N=0), lane 16 holds rows 8..15.
  if ((lane & 15) == 0) {
    int mb = half * 8;
#pragma unroll
    for (int i = 0; i < 8; ++i) {
      float v = acc[i];
      x_att[pbase + mb + i] = 1.0f / (1.0f + expf(-v));
    }
  }
#else
  if (lane < 16) {
    float s = 0.0f;
    for (int c = 0; c < C; ++c) s += xb[(size_t)c * NPTS + row] * att_w[c];
    x_att[pbase + row] = 1.0f / (1.0f + expf(-s));
  }
#endif
}

// flatten_x[g][c] = x[b][c][n] * x_att[g]   (gather table; L2-resident afterwards)
__global__ __launch_bounds__(256) void build_fx_kernel(const float* __restrict__ x,
                                                       const float* __restrict__ xatt,
                                                       float* __restrict__ fx) {
  int c = (int)blockIdx.y;
  int g = (int)(blockIdx.x * 256 + threadIdx.x);
  int b = g >> 14;
  int n = g & (NPTS - 1);
  fx[(size_t)g * C + c] = x[((size_t)b * C + c) * NPTS + n] * xatt[g];
}

// Stable top-256 per batch (iterative argmax extraction, ties -> lowest index).
// Destroys xatt (it is recomputed every launch). Writes initial flatten_cur.
__global__ __launch_bounds__(256) void topk_kernel(float* __restrict__ att,
                                                   int* __restrict__ fcur) {
  __shared__ float sv[256];
  __shared__ int si[256];
  int b = (int)blockIdx.x;
  float* a = att + (size_t)b * NPTS;
  for (int it = 0; it < CN; ++it) {
    float best = -3.4e38f; int bi = NPTS;
    for (int j = (int)threadIdx.x; j < NPTS; j += 256) {
      float v = a[j];
      if (v > best) { best = v; bi = j; }
    }
    sv[threadIdx.x] = best; si[threadIdx.x] = bi;
    __syncthreads();
    for (int s = 128; s > 0; s >>= 1) {
      if ((int)threadIdx.x < s) {
        float ov = sv[threadIdx.x + s]; int oi = si[threadIdx.x + s];
        if (ov > sv[threadIdx.x] || (ov == sv[threadIdx.x] && oi < si[threadIdx.x])) {
          sv[threadIdx.x] = ov; si[threadIdx.x] = oi;
        }
      }
      __syncthreads();
    }
    if (threadIdx.x == 0) {
      int w = si[0];
      fcur[b * CN + it] = b * NPTS + w;
      a[w] = -3.4e38f;
    }
    __syncthreads();
  }
}

// Persistent cooperative walk: one wave per curve (lane == neighbor k).
// 128 blocks x 512 threads = 2048 waves. Grid barriers for per-step global BN.
__global__ __launch_bounds__(512, 1) void walk_kernel(
    const float* __restrict__ fx, const int* __restrict__ idxb,
    const float* __restrict__ agent_w, const float* __restrict__ agent_g,
    const float* __restrict__ agent_bb, const float* __restrict__ mom_w,
    const float* __restrict__ mom_g, const float* __restrict__ mom_b,
    const int* __restrict__ fcur0, float* sbuf, float* lacc, float* macc,
    unsigned int* bar, float* __restrict__ out) {
  __shared__ __align__(16) float sh_cur[16][C];
  __shared__ __align__(16) float sh_pre[16][C];
  const int wiw = (int)(threadIdx.x >> 5);
  const int lane = (int)(threadIdx.x & 31);
  const int m = (int)blockIdx.x * 16 + wiw;   // curve id 0..2047
  const int b = m >> 8;
  const int n = m & 255;
  float* cur = sh_cur[wiw];
  float* pre = sh_pre[wiw];

  const float ag = agent_g[0], ab = agent_bb[0];
  const float mg0 = mom_g[0], mg1 = mom_g[1];
  const float mb0 = mom_b[0], mb1 = mom_b[1];
  const unsigned int nblk = gridDim.x;

  int fc = fcur0[m];
  if (lane < 16)
    ((float4*)pre)[lane] = ((const float4*)(fx + (size_t)fc * C))[lane];
  __syncthreads();

  unsigned int epoch = 0;

  for (int l = 0; l < CL; ++l) {
    if (l > 0) {
      // ---- momentum gate: mm = mom_w @ [cur;pre], BN over 2048 curves ----
      float pA = 0.0f, pB = 0.0f;
#pragma unroll
      for (int j = 0; j < 2; ++j) {
        int c = lane + j * 32;
        float cv = cur[c], pv = pre[c];
        pA += mom_w[c] * cv + mom_w[64 + c] * pv;
        pB += mom_w[128 + c] * cv + mom_w[192 + c] * pv;
      }
      float mm0 = wave_sum32(pA);
      float mm1 = wave_sum32(pB);
      if (lane == 0) {
        atomAddF(&macc[l * 4 + 0], mm0);
        atomAddF(&macc[l * 4 + 1], mm0 * mm0);
        atomAddF(&macc[l * 4 + 2], mm1);
        atomAddF(&macc[l * 4 + 3], mm1 * mm1);
      }
      grid_sync(bar, (++epoch) * nblk);
      const float invn = 1.0f / (float)NCURVE;
      float mu0 = ldAgent(&macc[l * 4 + 0]) * invn;
      float vr0 = ldAgent(&macc[l * 4 + 1]) * invn - mu0 * mu0;
      float mu1 = ldAgent(&macc[l * 4 + 2]) * invn;
      float vr1 = ldAgent(&macc[l * 4 + 3]) * invn - mu1 * mu1;
      float z0 = (mm0 - mu0) * rsqrtf(vr0 + EPS_BN) * mg0 + mb0;
      float z1 = (mm1 - mu1) * rsqrtf(vr1 + EPS_BN) * mg1 + mb1;
      float mx = fmaxf(z0, z1);
      float e0 = expf(z0 - mx), e1 = expf(z1 - mx);
      float r = 1.0f / (e0 + e1);
      if (lane == 0) {
        sbuf[b * 512 + n]       = e0 * r;
        sbuf[b * 512 + 256 + n] = e1 * r;
      }
      grid_sync(bar, (++epoch) * nblk);
      // torch .view reshape quirk: curve n blends with s at flat indices 2n, 2n+1
      int f0 = 2 * n, f1 = 2 * n + 1;
      float at0 = ldAgent(&sbuf[b * 512 + (f0 >> 8) * 256 + (f0 & 255)]);
      float at1 = ldAgent(&sbuf[b * 512 + (f1 >> 8) * 256 + (f1 & 255)]);
#pragma unroll
      for (int j = 0; j < 2; ++j) {
        int c = lane + j * 32;
        pre[c] = cur[c] * at0 + pre[c] * at1;
      }
      __syncthreads();
    }

    // ---- pick + fused accumulation: logits, crossover dot, norms ----
    int pk = idxb[(size_t)fc * KNB + lane] + b * NPTS;
    const float4* fr  = (const float4*)(fx + (size_t)pk * C);
    const float4* cu4 = (const float4*)cur;
    const float4* pr4 = (const float4*)pre;
    float accL = 0.0f, accD = 0.0f, accN = 0.0f, accA = 0.0f;
#pragma unroll
    for (int q = 0; q < 16; ++q) {
      float4 v  = fr[q];
      float4 cv = cu4[q];
      float4 pv = pr4[q];
      int c0 = q * 4;
      accL += v.x * agent_w[c0]     + pv.x * agent_w[64 + c0];
      accL += v.y * agent_w[c0 + 1] + pv.y * agent_w[64 + c0 + 1];
      accL += v.z * agent_w[c0 + 2] + pv.z * agent_w[64 + c0 + 2];
      accL += v.w * agent_w[c0 + 3] + pv.w * agent_w[64 + c0 + 3];
      if (l > 0) {
        float a0 = cv.x - pv.x, a1 = cv.y - pv.y, a2 = cv.z - pv.z, a3 = cv.w - pv.w;
        float d0 = v.x - cv.x,  d1 = v.y - cv.y,  d2 = v.z - cv.z,  d3 = v.w - cv.w;
        accD += a0 * d0 + a1 * d1 + a2 * d2 + a3 * d3;
        accN += d0 * d0 + d1 * d1 + d2 * d2 + d3 * d3;
        accA += a0 * a0 + a1 * a1 + a2 * a2 + a3 * a3;
      }
    }

    // ---- BN over all 65536 logits (global) ----
    float s1 = wave_sum32(accL);
    float s2 = wave_sum32(accL * accL);
    if (lane == 0) {
      atomAddF(&lacc[l * 2],     s1);
      atomAddF(&lacc[l * 2 + 1], s2);
    }
    grid_sync(bar, (++epoch) * nblk);
    const float invM = 1.0f / (float)(NCURVE * KNB);
    float mu = ldAgent(&lacc[l * 2]) * invM;
    float vr = ldAgent(&lacc[l * 2 + 1]) * invM - mu * mu;
    float lg = (accL - mu) * rsqrtf(vr + EPS_BN) * ag + ab;
    if (l > 0) {
      float div = fmaxf(sqrtf(accA) * sqrtf(accN), 1e-8f);
      float d = fminf(fmaxf(1.0f + accD / div, 0.0f), 1.0f);
      lg *= d;
    }

    // ---- straight-through forward == select argmax neighbor (ties->lowest k) ----
    float bv = lg; int bi = lane;
#pragma unroll
    for (int msk = 16; msk >= 1; msk >>= 1) {
      float ov = __shfl_xor(bv, msk, 32);
      int   oi = __shfl_xor(bi, msk, 32);
      if (ov > bv || (ov == bv && oi < bi)) { bv = ov; bi = oi; }
    }
    int sel = __shfl(pk, bi, 32);

    if (lane < 16) {
      float4 v = ((const float4*)(fx + (size_t)sel * C))[lane];
      ((float4*)cur)[lane] = v;
      size_t ob = (((size_t)b * C + lane * 4) * CN + n) * CL + (size_t)l;
      out[ob]                         = v.x;
      out[ob + (size_t)CN * CL]       = v.y;
      out[ob + (size_t)2 * CN * CL]   = v.z;
      out[ob + (size_t)3 * CN * CL]   = v.w;
    }
    __syncthreads();
    fc = sel;
  }
}

// ---------------- launch ----------------

extern "C" void kernel_launch(void* const* d_in, const int* in_sizes, int n_in,
                              void* d_out, int out_size, void* d_ws, size_t ws_size,
                              hipStream_t stream) {
  const float* x       = (const float*)d_in[0];
  // d_in[1] = xyz (unused by the reference forward)
  const int*   idx     = (const int*)d_in[2];
  const float* att_w   = (const float*)d_in[3];
  const float* agent_w = (const float*)d_in[4];
  const float* agent_g = (const float*)d_in[5];
  const float* agent_b = (const float*)d_in[6];
  const float* mom_w   = (const float*)d_in[7];
  const float* mom_g   = (const float*)d_in[8];
  const float* mom_b   = (const float*)d_in[9];
  float* out = (float*)d_out;

  char* w = (char*)d_ws;
  float* fx   = (float*)w;  w += sizeof(float) * (size_t)PTOT * C;
  float* xatt = (float*)w;  w += sizeof(float) * (size_t)PTOT;
  float* sbuf = (float*)w;  w += sizeof(float) * BN * 2 * CN;
  float* lacc = (float*)w;  w += sizeof(float) * CL * 2;
  float* macc = (float*)w;  w += sizeof(float) * CL * 4;
  int*   fcur = (int*)w;    w += sizeof(int) * NCURVE;
  unsigned int* bar = (unsigned int*)w;

  init_kernel<<<1, 256, 0, stream>>>(lacc, macc, bar);
  att_kernel<<<PTOT / 16 / 8, 256, 0, stream>>>(x, att_w, xatt);
  build_fx_kernel<<<dim3(PTOT / 256, C), 256, 0, stream>>>(x, xatt, fx);
  topk_kernel<<<BN, 256, 0, stream>>>(xatt, fcur);
  walk_kernel<<<NCURVE / 16, 512, 0, stream>>>(fx, idx, agent_w, agent_g, agent_b,
                                               mom_w, mom_g, mom_b, fcur, sbuf,
                                               lacc, macc, bar, out);
}
